// Colight_57123065037011
// MI455X (gfx1250) — matrix operand order repaired and checked by reading, not compiled, not run on hardware
//
#include <hip/hip_runtime.h>

typedef __attribute__((ext_vector_type(16))) _Float16 v16h;
typedef __attribute__((ext_vector_type(8)))  float    v8f;

#define B_    16
#define A_    859
#define N_    5
#define D_    32
#define RPB   (A_*N_)      // 4295 rows per batch for the adj GEMM
#define KPAD  864          // 859 padded to 27*32
#define NCH   27           // K chunks of 32

// ---------------------------------------------------------------------------
// MLP: x = relu(relu(agent@W1+b1)@W2+b2)   [13744,32]->[.,128]->[.,32]
// one wave per row, LDS-cooperative
// ---------------------------------------------------------------------------
__global__ void mlp_kernel(const float* __restrict__ agent,
                           const float* __restrict__ w1, const float* __restrict__ b1,
                           const float* __restrict__ w2, const float* __restrict__ b2,
                           float* __restrict__ x) {
  __shared__ float arow[4][32];
  __shared__ float hbuf[4][128];
  int wid = threadIdx.x >> 5;
  int l   = threadIdx.x & 31;
  int g   = blockIdx.x * 4 + wid;            // flat (b,a) row, 0..13743
  arow[wid][l] = agent[(size_t)g*32 + l];
  __syncthreads();
  #pragma unroll
  for (int t = 0; t < 4; ++t) {
    int j = l + 32*t;
    float acc = b1[j];
    #pragma unroll
    for (int k = 0; k < 32; ++k) acc += arow[wid][k] * w1[k*128 + j];
    hbuf[wid][j] = fmaxf(acc, 0.f);
  }
  __syncthreads();
  float acc = b2[l];
  for (int j = 0; j < 128; ++j) acc += hbuf[wid][j] * w2[j*32 + l];
  x[(size_t)g*32 + l] = fmaxf(acc, 0.f);
}

// ---------------------------------------------------------------------------
// Convert x [B,A,32] f32 -> f16 padded to [B,864,32] (pad rows = 0)
// ---------------------------------------------------------------------------
__global__ void cvt_kernel(const float* __restrict__ x, _Float16* __restrict__ xh) {
  int idx = blockIdx.x*256 + threadIdx.x;     // total B*KPAD*32 = 442368
  int b   = idx / (KPAD*32);
  int rem = idx - b*KPAD*32;
  int row = rem >> 5;
  int k   = rem & 31;
  float v = (row < A_) ? x[((size_t)b*A_ + row)*32 + k] : 0.f;
  xh[idx] = (_Float16)v;
}

// ---------------------------------------------------------------------------
// nbr = adj @ x  per batch: [4295,859]x[859,32], f16 WMMA with f32 accum.
// One wave per 16x32 output tile; adj streamed f32 (coalesced) -> LDS ->
// f16 A-operand (ds_load_b64 + v_cvt_pk_f16_f32).
// A layout (16-bit 16x32): lane 0-15 = M, K pairs {0..7,16..23};
//                          lane 16-31 = same M, K pairs {8..15,24..31}.
// B layout (16-bit 32x16): lane%16 = N column, element s = K (same order as A).
// Guards are split by uniformity: row guard is scalar-uniform, K guard only
// exists in the final chunk and is a branchless clamp+cndmask.
// ---------------------------------------------------------------------------
__global__ void __launch_bounds__(32) nbr_wmma_kernel(
    const float* __restrict__ adj, const _Float16* __restrict__ xh,
    float* __restrict__ nbr) {
  __shared__ float atile[16*32];
  int t   = blockIdx.x;                      // row tile 0..268
  int b   = blockIdx.y;                      // batch
  int l   = threadIdx.x;
  int sel = l >> 4;
  int ml  = l & 15;
  v8f c0 = {}; v8f c1 = {};
  const float*    arow0 = adj + (size_t)b*RPB*A_ + (size_t)t*16*A_;
  const _Float16* xrow  = xh + (size_t)b*KPAD*32;
  const bool fullRows  = (t*16 + 16) <= RPB;            // uniform per WG
  const int  validRows = fullRows ? 16 : (RPB - t*16);  // uniform per WG

  auto chunk = [&](int kbase, bool kEdge) {
    int  kk  = kbase + l;
    bool kOk = !kEdge || (kk < A_);
    int  kcl = kOk ? kk : (A_ - 1);          // clamped, always in-bounds
    #pragma unroll
    for (int m = 0; m < 16; ++m) {
      float v = 0.f;
      if (m < validRows) {                   // scalar-uniform branch
        v = arow0[(size_t)m*A_ + kcl];
        if (!kOk) v = 0.f;                   // single cndmask (edge chunk only)
      }
      atile[m*32 + l] = v;
    }
    __syncthreads();
    v16h av;
    const float2* ap = reinterpret_cast<const float2*>(atile + ml*32);
    #pragma unroll
    for (int p = 0; p < 8; ++p) {
      int kpair = (sel == 0) ? ((p < 4) ? p      : 8 + (p-4))
                             : ((p < 4) ? 4 + p  : 12 + (p-4));
      float2 u = ap[kpair];                  // ds_load_b64, 8B aligned
      av[2*p]   = (_Float16)u.x;
      av[2*p+1] = (_Float16)u.y;
    }
    v16h bva, bvb;
    #pragma unroll
    for (int s = 0; s < 16; ++s) {
      int kb = kbase + ((s < 8) ? (8*sel + s) : (16 + 8*sel + (s - 8)));
      bva[s] = xrow[(size_t)kb*32 + ml];       // N cols 0..15
      bvb[s] = xrow[(size_t)kb*32 + 16 + ml];  // N cols 16..31
    }
    c0 = __builtin_amdgcn_wmma_f32_16x16x32_f16(false, av, false, bva, (short)0, c0, false, false);
    c1 = __builtin_amdgcn_wmma_f32_16x16x32_f16(false, av, false, bvb, (short)0, c1, false, false);
    __syncthreads();
  };

  for (int ch = 0; ch < NCH - 1; ++ch) chunk(ch*32, false);  // clean path
  chunk((NCH-1)*32, true);                                   // K-edge path

  // C/D layout: VGPR v, lane L: M = v + 8*(L>>4), N = L&15
  int col   = l & 15;
  int rhalf = 8*(l >> 4);
  if (fullRows) {
    #pragma unroll
    for (int v = 0; v < 8; ++v) {
      size_t o = ((size_t)b*RPB + t*16 + v + rhalf)*32;
      nbr[o + col]      = c0[v];
      nbr[o + 16 + col] = c1[v];
    }
  } else {
    #pragma unroll
    for (int v = 0; v < 8; ++v) {
      int r = t*16 + v + rhalf;
      if (r < RPB) {
        size_t o = ((size_t)b*RPB + r)*32;
        nbr[o + col]      = c0[v];
        nbr[o + 16 + col] = c1[v];
      }
    }
  }
}

// ---------------------------------------------------------------------------
// ah = relu(x@ah_w+b) -> [nv,dv]; nr = relu(nbr@nr_w+b) -> [nv,N,dv];
// scores[h,e] = <ah[h,:], nr[h,e,:]>.  One wave per agent.
// ---------------------------------------------------------------------------
__global__ void heads_kernel(const float* __restrict__ x, const float* __restrict__ nbr,
                             const float* __restrict__ ah_w, const float* __restrict__ ah_b,
                             const float* __restrict__ nr_w, const float* __restrict__ nr_b,
                             float* __restrict__ scores) {
  __shared__ float xrow[4][32];
  __shared__ float nb[4][160];
  __shared__ float ah[4][128];
  __shared__ float nr[4][640];
  int wid = threadIdx.x >> 5;
  int l   = threadIdx.x & 31;
  int g   = blockIdx.x*4 + wid;
  xrow[wid][l] = x[(size_t)g*32 + l];
  for (int i = l; i < 160; i += 32) nb[wid][i] = nbr[(size_t)g*160 + i];
  __syncthreads();
  #pragma unroll
  for (int tt = 0; tt < 4; ++tt) {
    int j = l + 32*tt;
    float acc = ah_b[j];
    #pragma unroll
    for (int k = 0; k < 32; ++k) acc += xrow[wid][k]*ah_w[k*128+j];
    ah[wid][j] = fmaxf(acc, 0.f);
  }
  for (int e = 0; e < 5; ++e) {
    #pragma unroll
    for (int tt = 0; tt < 4; ++tt) {
      int j = l + 32*tt;
      float acc = nr_b[j];
      #pragma unroll
      for (int k = 0; k < 32; ++k) acc += nb[wid][e*32+k]*nr_w[k*128+j];
      nr[wid][e*128+j] = fmaxf(acc, 0.f);
    }
  }
  __syncthreads();
  for (int s = l; s < 40; s += 32) {         // s = h*5+e
    int h = s / 5, e = s % 5;
    float sc = 0.f;
    #pragma unroll
    for (int d = 0; d < 16; ++d) sc += ah[wid][d*8+h]*nr[wid][e*128 + d*8+h];
    scores[(size_t)g*40 + s] = sc;
  }
}

// ---------------------------------------------------------------------------
// softmax over the AGENT axis (axis=1): one block per (b,h,e); writes att
// directly into d_out's att_record region.
// ---------------------------------------------------------------------------
__global__ void softmax_kernel(const float* __restrict__ scores, float* __restrict__ out,
                               int layer) {
  __shared__ float red[256];
  int id  = blockIdx.x;                      // b*40 + (h*5+e)
  int b   = id / 40;
  int s   = id % 40;
  int tid = threadIdx.x;
  float m = -3.4e38f;
  for (int a = tid; a < A_; a += 256)
    m = fmaxf(m, scores[((size_t)b*A_ + a)*40 + s]);
  red[tid] = m; __syncthreads();
  for (int st = 128; st > 0; st >>= 1) { if (tid < st) red[tid] = fmaxf(red[tid], red[tid+st]); __syncthreads(); }
  m = red[0]; __syncthreads();
  float sum = 0.f;
  for (int a = tid; a < A_; a += 256)
    sum += __expf(scores[((size_t)b*A_ + a)*40 + s] - m);
  red[tid] = sum; __syncthreads();
  for (int st = 128; st > 0; st >>= 1) { if (tid < st) red[tid] += red[tid+st]; __syncthreads(); }
  float inv = 1.f / red[0];
  for (int a = tid; a < A_; a += 256) {
    float v = __expf(scores[((size_t)b*A_ + a)*40 + s] - m) * inv;
    out[128 + ((size_t)((b*2 + layer)*A_ + a))*40 + s] = v;   // [B,2,A,8,5]
  }
}

// ---------------------------------------------------------------------------
// out16[d] = sum_{h,e} att[h,e]*relu(nbr[e]@nh_w+b)[d*8+h];  x' = relu(out16/8 @ o_w + o_b)
// One wave per agent; nh recomputed on the fly; LDS atomic accumulation.
// ---------------------------------------------------------------------------
__global__ void combine_kernel(const float* __restrict__ nbr, const float* __restrict__ att_out,
                               const float* __restrict__ nh_w, const float* __restrict__ nh_b,
                               const float* __restrict__ o_w, const float* __restrict__ o_b,
                               float* __restrict__ xn, int layer) {
  __shared__ float nb[4][160];
  __shared__ float at[4][40];
  __shared__ float o16[4][16];
  int wid = threadIdx.x >> 5;
  int l   = threadIdx.x & 31;
  int g   = blockIdx.x*4 + wid;
  int b   = g / A_;
  int a   = g - b*A_;
  for (int i = l; i < 160; i += 32) nb[wid][i] = nbr[(size_t)g*160 + i];
  size_t attBase = 128 + ((size_t)((b*2 + layer)*A_ + a))*40;
  for (int i = l; i < 40; i += 32) at[wid][i] = att_out[attBase + i];
  if (l < 16) o16[wid][l] = 0.f;
  __syncthreads();
  for (int e = 0; e < 5; ++e) {
    #pragma unroll
    for (int tt = 0; tt < 4; ++tt) {
      int j = l + 32*tt;
      float acc = nh_b[j];
      #pragma unroll
      for (int k = 0; k < 32; ++k) acc += nb[wid][e*32+k]*nh_w[k*128+j];
      acc = fmaxf(acc, 0.f);
      int d = j >> 3, h = j & 7;
      atomicAdd(&o16[wid][d], at[wid][h*5+e]*acc);
    }
  }
  __syncthreads();
  float acc = o_b[l];
  #pragma unroll
  for (int d = 0; d < 16; ++d) acc += (o16[wid][d]*0.125f)*o_w[d*32 + l];
  xn[(size_t)g*32 + l] = fmaxf(acc, 0.f);
}

// ---------------------------------------------------------------------------
// q = (x @ fin_w + fin_b)[:,0,:]  -> d_out[0..127]
// ---------------------------------------------------------------------------
__global__ void final_kernel(const float* __restrict__ x, const float* __restrict__ fw,
                             const float* __restrict__ fb, float* __restrict__ out) {
  int tid = threadIdx.x;                     // 128 = B*ACT
  int b = tid >> 3, c = tid & 7;
  const float* xr = x + (size_t)b*A_*32;     // agent 0
  float acc = fb[c];
  #pragma unroll
  for (int d = 0; d < 32; ++d) acc += xr[d]*fw[d*8+c];
  out[tid] = acc;
}

// ---------------------------------------------------------------------------
extern "C" void kernel_launch(void* const* d_in, const int* in_sizes, int n_in,
                              void* d_out, int out_size, void* d_ws, size_t ws_size,
                              hipStream_t stream) {
  const float* agent = (const float*)d_in[0];
  const float* adj   = (const float*)d_in[1];
  const float* w1 = (const float*)d_in[2]; const float* b1 = (const float*)d_in[3];
  const float* w2 = (const float*)d_in[4]; const float* b2 = (const float*)d_in[5];
  const float* fw = (const float*)d_in[6]; const float* fb = (const float*)d_in[7];

  char* ws = (char*)d_ws;
  size_t off = 0;
  auto alloc = [&](size_t bytes) { void* p = ws + off; off = (off + bytes + 255) & ~(size_t)255; return p; };
  float*    xA     = (float*)   alloc((size_t)B_*A_*32*4);
  float*    xB     = (float*)   alloc((size_t)B_*A_*32*4);
  _Float16* xh     = (_Float16*)alloc((size_t)B_*KPAD*32*2);
  float*    nbr    = (float*)   alloc((size_t)B_*RPB*32*4);
  float*    scores = (float*)   alloc((size_t)B_*A_*40*4);
  float*    out    = (float*)d_out;

  mlp_kernel<<<(B_*A_)/4, 128, 0, stream>>>(agent, w1, b1, w2, b2, xA);

  float* xc = xA; float* xn = xB;
  for (int layer = 0; layer < 2; ++layer) {
    const float* ah_w = (const float*)d_in[8 + layer*8 + 0];
    const float* ah_b = (const float*)d_in[8 + layer*8 + 1];
    const float* nr_w = (const float*)d_in[8 + layer*8 + 2];
    const float* nr_b = (const float*)d_in[8 + layer*8 + 3];
    const float* nh_w = (const float*)d_in[8 + layer*8 + 4];
    const float* nh_b = (const float*)d_in[8 + layer*8 + 5];
    const float* o_w  = (const float*)d_in[8 + layer*8 + 6];
    const float* o_b  = (const float*)d_in[8 + layer*8 + 7];

    cvt_kernel<<<(B_*KPAD*32)/256, 256, 0, stream>>>(xc, xh);
    dim3 g((RPB + 15)/16, B_);
    nbr_wmma_kernel<<<g, 32, 0, stream>>>(adj, xh, nbr);
    heads_kernel<<<(B_*A_)/4, 128, 0, stream>>>(xc, nbr, ah_w, ah_b, nr_w, nr_b, scores);
    softmax_kernel<<<B_*40, 256, 0, stream>>>(scores, out, layer);
    combine_kernel<<<(B_*A_)/4, 128, 0, stream>>>(nbr, out, nh_w, nh_b, o_w, o_b, xn, layer);
    float* tmp = xc; xc = xn; xn = tmp;
  }
  final_kernel<<<1, 128, 0, stream>>>(xc, fw, fb, out);
}
